// DependencyParser_18322330485474
// MI455X (gfx1250) — compile-verified, matrix operand order checked by reading, not run on hardware
//
#include <hip/hip_runtime.h>
#include <hip/hip_bf16.h>
#include <math.h>

// ---------------------------------------------------------------------------
// Types for CDNA5 WMMA (wave32, 16x16x32 f16 -> f32 accumulate)
// ---------------------------------------------------------------------------
typedef __attribute__((ext_vector_type(16))) _Float16 v16h;
typedef __attribute__((ext_vector_type(8)))  _Float16 v8h;
typedef __attribute__((ext_vector_type(8)))  float    v8f;

#define T_SEQ   256
#define HID     400
#define G4      1600          // 4*HID
#define KP0     416           // 400 padded to multiple of 32
#define K1      800           // 2*HID (already multiple of 32)
#define NP_FC   112           // 100 padded to multiple of 16
#define H2      800           // 2*HID concat width

// ---------------------------------------------------------------------------
// 1) Embedding gather + concat -> padded f16 [256][416]
// ---------------------------------------------------------------------------
__global__ void embed_concat_f16(const int* __restrict__ widx,
                                 const int* __restrict__ pidx,
                                 const float* __restrict__ wemb,  // [50000][300]
                                 const float* __restrict__ pemb,  // [50][100]
                                 _Float16* __restrict__ xb)       // [256][416]
{
    int t = blockIdx.x;          // 0..255
    int j = threadIdx.x;         // 0..415
    float v;
    if (j < 300)      v = wemb[(long)widx[t] * 300 + j];
    else if (j < 400) v = pemb[(long)pidx[t] * 100 + (j - 300)];
    else              v = 0.0f;
    xb[t * KP0 + j] = (_Float16)v;
}

// ---------------------------------------------------------------------------
// 2) Generic fp32 -> f16 convert with row/col padding and strided source rows
// ---------------------------------------------------------------------------
__global__ void cvt_pad_f16(const float* __restrict__ src, _Float16* __restrict__ dst,
                            int N, int K, int srcStride, int srcColOff,
                            int Npad, int Kpad)
{
    int idx = blockIdx.x * blockDim.x + threadIdx.x;
    if (idx >= Npad * Kpad) return;
    int n = idx / Kpad, k = idx % Kpad;
    float v = (n < N && k < K) ? src[(long)n * srcStride + srcColOff + k] : 0.0f;
    dst[idx] = (_Float16)v;
}

// ---------------------------------------------------------------------------
// 3) WMMA GEMM:  C[M][N] = A[M][K] * B[N][K]^T   (both row-major, K%32==0,
//    M%64==0).  One wave computes FOUR 16x16 C tiles stacked in M, sharing a
//    single B fragment per K-slab (4 independent WMMAs per B load -> better
//    wmma:vmem ratio and no D->A/B hazards between them).
//    Fragment layout per ISA 7.12.2 (16-bit A 16x32):
//      lanes 0-15  : row = lane,    K = k0+0..7  and k0+16..23
//      lanes 16-31 : row = lane-16, K = k0+8..15 and k0+24..31
//    C layout: VGPR v, lanes 0-15 -> M=v, N=lane; lanes 16-31 -> M=v+8.
// ---------------------------------------------------------------------------
__device__ __forceinline__ v16h load_frag(const _Float16* p) {
    v8h lo = *(const v8h*)(p);
    v8h hi = *(const v8h*)(p + 16);
    v16h f;
#pragma unroll
    for (int i = 0; i < 8; ++i) { f[i] = lo[i]; f[i + 8] = hi[i]; }
    return f;
}

__global__ void wmma_gemm_nt4(const _Float16* __restrict__ A,  // [M][K]
                              const _Float16* __restrict__ B,  // [N][K]
                              float* __restrict__ C,           // [M][N]
                              int M, int N, int K)
{
    const int lane  = threadIdx.x & 31;
    const int wave  = threadIdx.x >> 5;
    const int ntile = blockIdx.x * (blockDim.x >> 5) + wave;   // wave-uniform
    const int mbase = blockIdx.y * 64;                         // 4 m-tiles
    if (ntile * 16 >= N) return;          // uniform per wave: EXEC stays all-1s

    const int half = lane >> 4;
    const int r    = lane & 15;
    const _Float16* brow = B + (long)(ntile * 16 + r) * K;
    const _Float16* arow = A + (long)(mbase + r) * K;

    v8f acc0 = {}, acc1 = {}, acc2 = {}, acc3 = {};
    for (int k0 = 0; k0 < K; k0 += 32) {
        if (k0 + 32 < K) {                 // uniform branch
            __builtin_prefetch(brow + k0 + 32, 0, 3);   // WGP-scope prefetch
            __builtin_prefetch(arow + k0 + 32, 0, 3);
        }
        const int ka = k0 + half * 8;
        v16h b  = load_frag(brow + ka);
        v16h a0 = load_frag(arow + ka);
        v16h a1 = load_frag(arow + (long)16 * K + ka);
        v16h a2 = load_frag(arow + (long)32 * K + ka);
        v16h a3 = load_frag(arow + (long)48 * K + ka);
        acc0 = __builtin_amdgcn_wmma_f32_16x16x32_f16(false, a0, false, b, (short)0, acc0, false, false);
        acc1 = __builtin_amdgcn_wmma_f32_16x16x32_f16(false, a1, false, b, (short)0, acc1, false, false);
        acc2 = __builtin_amdgcn_wmma_f32_16x16x32_f16(false, a2, false, b, (short)0, acc2, false, false);
        acc3 = __builtin_amdgcn_wmma_f32_16x16x32_f16(false, a3, false, b, (short)0, acc3, false, false);
    }

    float* cbase = C + (long)(mbase + half * 8) * N + ntile * 16 + r;
#pragma unroll
    for (int v = 0; v < 8; ++v) cbase[(long)(v)      * N] = acc0[v];
#pragma unroll
    for (int v = 0; v < 8; ++v) cbase[(long)(v + 16) * N] = acc1[v];
#pragma unroll
    for (int v = 0; v < 8; ++v) cbase[(long)(v + 32) * N] = acc2[v];
#pragma unroll
    for (int v = 0; v < 8; ++v) cbase[(long)(v + 48) * N] = acc3[v];
}

// ---------------------------------------------------------------------------
// 4) LSTM recurrence, one direction per workgroup (gridDim.x == 2).
//    xg already holds x @ Wih^T; biases bih+bhh added here.  h,c,gates in LDS.
//    Gate order i,f,g,o (torch), H=400 each.  dir1 runs time reversed but
//    writes outputs at original time index (matches lax.scan(reverse=True)).
// ---------------------------------------------------------------------------
__global__ __launch_bounds__(1024)
void lstm_recurrent(const float* __restrict__ xg,    // [2][256][1600]
                    const float* __restrict__ Whh,   // [2][1600][400]
                    const float* __restrict__ bih,   // [2][1600]
                    const float* __restrict__ bhh,   // [2][1600]
                    _Float16* __restrict__ houtH)    // [256][800] (f16 for next GEMM)
{
    const int dir = blockIdx.x;
    const int tid = threadIdx.x;                     // 0..1023
    __shared__ float hs[HID];
    __shared__ float cs[HID];
    __shared__ float gs[G4];

    for (int i = tid; i < HID; i += 1024) { hs[i] = 0.0f; cs[i] = 0.0f; }
    __syncthreads();

    const float* xgd = xg  + (long)dir * T_SEQ * G4;
    const float* Wd  = Whh + (long)dir * G4 * HID;
    const float* b1  = bih + dir * G4;
    const float* b2  = bhh + dir * G4;

    for (int s = 0; s < T_SEQ; ++s) {
        const int t = dir ? (T_SEQ - 1 - s) : s;
        // gates = xg[t] + b + h @ Whh^T
        for (int g = tid; g < G4; g += 1024) {
            const float* w = Wd + (long)g * HID;
            float acc = xgd[(long)t * G4 + g] + b1[g] + b2[g];
#pragma unroll 4
            for (int k = 0; k < HID; ++k) acc = fmaf(w[k], hs[k], acc);
            gs[g] = acc;
        }
        __syncthreads();
        if (tid < HID) {
            float gi = 1.0f / (1.0f + __expf(-gs[tid]));
            float gf = 1.0f / (1.0f + __expf(-gs[HID + tid]));
            float gg = tanhf(gs[2 * HID + tid]);
            float go = 1.0f / (1.0f + __expf(-gs[3 * HID + tid]));
            float c  = gf * cs[tid] + gi * gg;
            float h  = go * tanhf(c);
            cs[tid] = c;
            hs[tid] = h;
            houtH[(long)t * H2 + dir * HID + tid] = (_Float16)h;
        }
        __syncthreads();
    }
}

// ---------------------------------------------------------------------------
// 5) Edge scorer using factorized projections:
//    score(h,m) = fc2_b + sum_j fc2_W[j] * tanh(P1[h][j] + P2[m][j] + fc1_b[j])
//    Output index follows _edge_pairs ordering (h-major, m ascending,
//    skipping m==0 and m==h).
// ---------------------------------------------------------------------------
__global__ void edge_score(const float* __restrict__ P1,    // [256][112]
                           const float* __restrict__ P2,    // [256][112]
                           const float* __restrict__ fc1_b, // [100]
                           const float* __restrict__ fc2_W, // [100]
                           const float* __restrict__ fc2_b, // [1]
                           float* __restrict__ out)         // [65025]
{
    const int h = blockIdx.x;     // 0..255
    const int m = threadIdx.x;    // 0..255
    if (m == 0 || m == h) return;
    const int base = (h == 0) ? 0 : 255 + (h - 1) * 254;
    const int pos  = (m - 1) - ((h >= 1 && m > h) ? 1 : 0);
    const float* p1 = P1 + h * NP_FC;
    const float* p2 = P2 + m * NP_FC;
    float acc = fc2_b[0];
#pragma unroll 4
    for (int j = 0; j < 100; ++j)
        acc = fmaf(fc2_W[j], tanhf(p1[j] + p2[j] + fc1_b[j]), acc);
    out[base + pos] = acc;
}

// ---------------------------------------------------------------------------
// Host side
// ---------------------------------------------------------------------------
extern "C" void kernel_launch(void* const* d_in, const int* in_sizes, int n_in,
                              void* d_out, int out_size, void* d_ws, size_t ws_size,
                              hipStream_t stream) {
    const int*   word_idx = (const int*)  d_in[0];
    const int*   pos_idx  = (const int*)  d_in[1];
    const float* word_emb = (const float*)d_in[2];
    const float* pos_emb  = (const float*)d_in[3];
    const float* Wih0     = (const float*)d_in[4];   // [2][1600][400]
    const float* Whh0     = (const float*)d_in[5];   // [2][1600][400]
    const float* bih0     = (const float*)d_in[6];
    const float* bhh0     = (const float*)d_in[7];
    const float* Wih1     = (const float*)d_in[8];   // [2][1600][800]
    const float* Whh1     = (const float*)d_in[9];   // [2][1600][400]
    const float* bih1     = (const float*)d_in[10];
    const float* bhh1     = (const float*)d_in[11];
    const float* fc1_W    = (const float*)d_in[12];  // [100][1600]
    const float* fc1_b    = (const float*)d_in[13];
    const float* fc2_W    = (const float*)d_in[14];
    const float* fc2_b    = (const float*)d_in[15];
    float* out            = (float*)d_out;

    // Workspace layout (bytes, 256-aligned)
    char* ws = (char*)d_ws;
    _Float16* xb     = (_Float16*)(ws + 0);          // [256][416]        212992 B
    _Float16* Wih0b  = (_Float16*)(ws + 212992);     // [3200][416]      2662400 B
    float*    xg0    = (float*)   (ws + 2875392);    // [2][256][1600]   3276800 B
    _Float16* h0h    = (_Float16*)(ws + 6152192);    // [256][800]        409600 B
    _Float16* Wih1b  = (_Float16*)(ws + 6561792);    // [3200][800]      5120000 B
    float*    xg1    = (float*)   (ws + 11681792);   // [2][256][1600]   3276800 B
    _Float16* h1h    = (_Float16*)(ws + 14958592);   // [256][800]        409600 B
    _Float16* W1h    = (_Float16*)(ws + 15368192);   // [112][800]        179200 B
    _Float16* W2h    = (_Float16*)(ws + 15547392);   // [112][800]        179200 B
    float*    P1     = (float*)   (ws + 15726592);   // [256][112]        114688 B
    float*    P2     = (float*)   (ws + 15841280);   // [256][112]        114688 B
    // total ~15.96 MB

    // 1) embeddings -> f16 padded input
    embed_concat_f16<<<T_SEQ, KP0, 0, stream>>>(word_idx, pos_idx, word_emb, pos_emb, xb);

    // 2) weight conversions to f16 (with K padding where needed)
    {
        int n = 3200 * KP0;   // Wih0 both dirs as [3200][400] -> [3200][416]
        cvt_pad_f16<<<(n + 255) / 256, 256, 0, stream>>>(Wih0, Wih0b, 3200, 400, 400, 0, 3200, KP0);
        n = 3200 * K1;        // Wih1 both dirs [3200][800] -> [3200][800]
        cvt_pad_f16<<<(n + 255) / 256, 256, 0, stream>>>(Wih1, Wih1b, 3200, 800, 800, 0, 3200, K1);
        n = NP_FC * K1;       // fc1 left half (cols 0..799), rows padded 100->112
        cvt_pad_f16<<<(n + 255) / 256, 256, 0, stream>>>(fc1_W, W1h, 100, 800, 1600, 0,   NP_FC, K1);
        cvt_pad_f16<<<(n + 255) / 256, 256, 0, stream>>>(fc1_W, W2h, 100, 800, 1600, 800, NP_FC, K1);
    }

    // 3) layer-0 input projections: xg0[d] = xb @ Wih0[d]^T   (M=256,N=1600,K=416)
    {
        dim3 grid(25, 4), block(128);   // 4 waves/block * 25 = 100 n-tiles; 4 m-groups of 64
        wmma_gemm_nt4<<<grid, block, 0, stream>>>(xb, Wih0b,               xg0,              T_SEQ, G4, KP0);
        wmma_gemm_nt4<<<grid, block, 0, stream>>>(xb, Wih0b + 1600 * KP0,  xg0 + T_SEQ * G4, T_SEQ, G4, KP0);
    }

    // 4) layer-0 recurrence (both directions concurrently, one WGP each)
    lstm_recurrent<<<2, 1024, 0, stream>>>(xg0, Whh0, bih0, bhh0, h0h);

    // 5) layer-1 input projections: xg1[d] = h0 @ Wih1[d]^T   (M=256,N=1600,K=800)
    {
        dim3 grid(25, 4), block(128);
        wmma_gemm_nt4<<<grid, block, 0, stream>>>(h0h, Wih1b,              xg1,              T_SEQ, G4, K1);
        wmma_gemm_nt4<<<grid, block, 0, stream>>>(h0h, Wih1b + 1600 * K1,  xg1 + T_SEQ * G4, T_SEQ, G4, K1);
    }

    // 6) layer-1 recurrence
    lstm_recurrent<<<2, 1024, 0, stream>>>(xg1, Whh1, bih1, bhh1, h1h);

    // 7) factorized edge-scorer projections: P1 = h1 @ fc1_W[:, :800]^T, P2 = ...[:,800:]
    {
        dim3 grid(7, 4), block(32);     // 7 n-tiles (112/16), 4 m-groups of 64
        wmma_gemm_nt4<<<grid, block, 0, stream>>>(h1h, W1h, P1, T_SEQ, NP_FC, K1);
        wmma_gemm_nt4<<<grid, block, 0, stream>>>(h1h, W2h, P2, T_SEQ, NP_FC, K1);
    }

    // 8) per-edge tanh-MLP head -> [65025] scores in _edge_pairs order
    edge_score<<<T_SEQ, T_SEQ, 0, stream>>>(P1, P2, fc1_b, fc2_W, fc2_b, out);
}
